// MultiHeadAttention_30691836297580
// MI455X (gfx1250) — matrix operand order repaired and checked by reading, not compiled, not run on hardware
//
#include <hip/hip_runtime.h>

// ---------------- types ----------------
typedef __bf16 bf16;
typedef __attribute__((ext_vector_type(8)))  __bf16 v8bf;
typedef __attribute__((ext_vector_type(16))) __bf16 v16bf;
typedef __attribute__((ext_vector_type(8)))  float  v8f;

#define B_  2
#define S_  2048
#define E_  1024
#define H_  16
#define DK_ 64

static __device__ inline v16bf mk16(v8bf lo, v8bf hi) {
  v16bf r;
#pragma unroll
  for (int i = 0; i < 8; ++i) { r[i] = lo[i]; r[i + 8] = hi[i]; }
  return r;
}

static __device__ inline v8f zero8() {
  v8f z;
#pragma unroll
  for (int i = 0; i < 8; ++i) z[i] = 0.0f;
  return z;
}

// ---------------- conversion kernels ----------------
__global__ void __launch_bounds__(256) cvt_f32_bf16_k(const float* __restrict__ in,
                                                      bf16* __restrict__ out, int n) {
  int i = blockIdx.x * blockDim.x + threadIdx.x;
  if (i < n) out[i] = (bf16)in[i];
}

// Wt[o*E + i] = W[i*E + o]  (W stored [in,out]; Wt stored [out,in])
__global__ void __launch_bounds__(256) cvt_wT_k(const float* __restrict__ W,
                                                bf16* __restrict__ Wt) {
  int idx = blockIdx.x * blockDim.x + threadIdx.x;   // over E*E
  int o = idx >> 10;
  int i = idx & (E_ - 1);
  Wt[idx] = (bf16)W[i * E_ + o];
}

// ---------------- fused QKV projection ----------------
// Each wave computes one 32x32 output tile (2x2 WMMA fragments) of one of Q/K/V.
// tiles: 128 (M/32) * 32 (E/32) * 3 = 12288 ; 8 waves/block -> 1536 blocks
__global__ void __launch_bounds__(256)
qkv_gemm_k(const bf16* __restrict__ xb,
           const bf16* __restrict__ wqT, const bf16* __restrict__ wkT, const bf16* __restrict__ wvT,
           const float* __restrict__ bq, const float* __restrict__ bk, const float* __restrict__ bv,
           bf16* __restrict__ Qb, bf16* __restrict__ Kb, bf16* __restrict__ Vt) {
  int wid  = threadIdx.x >> 5;
  int lane = threadIdx.x & 31;
  int half = lane >> 4;
  int n    = lane & 15;

  int tile   = blockIdx.x * 8 + wid;
  int mtile  = tile & 127;         // 128 m-tiles of 32 rows
  int nt_all = tile >> 7;          // 0..95
  int p      = nt_all >> 5;        // 0=Q 1=K 2=V
  int ntile  = nt_all & 31;        // 32 n-tiles of 32 cols

  const bf16* wt = (p == 0) ? wqT : (p == 1) ? wkT : wvT;
  const bf16* arow0 = xb + (size_t)(mtile * 32 + n) * E_;        // A rows m, m+16
  const bf16* arow1 = arow0 + (size_t)16 * E_;
  const bf16* brow0 = wt + (size_t)(ntile * 32 + n) * E_;        // B cols n, n+16
  const bf16* brow1 = brow0 + (size_t)16 * E_;

  v8f acc[2][2];
#pragma unroll
  for (int mi = 0; mi < 2; ++mi)
#pragma unroll
    for (int ni = 0; ni < 2; ++ni) acc[mi][ni] = zero8();

  for (int k = 0; k < E_; k += 32) {
    v16bf a0 = mk16(*(const v8bf*)(arow0 + k + half * 8),
                    *(const v8bf*)(arow0 + k + 16 + half * 8));
    v16bf a1 = mk16(*(const v8bf*)(arow1 + k + half * 8),
                    *(const v8bf*)(arow1 + k + 16 + half * 8));
    v16bf b0 = mk16(*(const v8bf*)(brow0 + k + half * 8),
                    *(const v8bf*)(brow0 + k + 16 + half * 8));
    v16bf b1 = mk16(*(const v8bf*)(brow1 + k + half * 8),
                    *(const v8bf*)(brow1 + k + 16 + half * 8));
    acc[0][0] = __builtin_amdgcn_wmma_f32_16x16x32_bf16(false, a0, false, b0, (short)0, acc[0][0], false, false);
    acc[0][1] = __builtin_amdgcn_wmma_f32_16x16x32_bf16(false, a0, false, b1, (short)0, acc[0][1], false, false);
    acc[1][0] = __builtin_amdgcn_wmma_f32_16x16x32_bf16(false, a1, false, b0, (short)0, acc[1][0], false, false);
    acc[1][1] = __builtin_amdgcn_wmma_f32_16x16x32_bf16(false, a1, false, b1, (short)0, acc[1][1], false, false);
  }

  const float* bias = (p == 0) ? bq : (p == 1) ? bk : bv;

#pragma unroll
  for (int ni = 0; ni < 2; ++ni) {
    int col = ntile * 32 + ni * 16 + n;   // 0..1023
    int h   = col >> 6;
    int dk  = col & 63;
    float bvl = bias[col];
#pragma unroll
    for (int mi = 0; mi < 2; ++mi) {
#pragma unroll
      for (int r = 0; r < 8; ++r) {
        int m  = half * 8 + r;
        int sg = mtile * 32 + mi * 16 + m;   // global row in [0,4096)
        int bb = sg >> 11;
        int ss = sg & (S_ - 1);
        bf16 val = (bf16)(acc[mi][ni][r] + bvl);
        if (p == 0)
          Qb[(((size_t)(bb * H_ + h)) * S_ + ss) * DK_ + dk] = val;
        else if (p == 1)
          Kb[(((size_t)(bb * H_ + h)) * S_ + ss) * DK_ + dk] = val;
        else
          Vt[(((size_t)(bb * H_ + h)) * DK_ + dk) * S_ + ss] = val;   // V transposed
      }
    }
  }
}

// ---------------- flash attention ----------------
// grid = (S/16/8, B*H), 256 threads = 8 waves; wave -> one 16-row q tile.
__global__ void __launch_bounds__(256)
attn_k(const bf16* __restrict__ Qb, const bf16* __restrict__ Kb,
       const bf16* __restrict__ Vt, bf16* __restrict__ ctxb) {
  __shared__ bf16 pbuf[8][16 * 40];   // per-wave 16x32 P tile, row stride 40 (16B aligned)

  int wid  = threadIdx.x >> 5;
  int lane = threadIdx.x & 31;
  int half = lane >> 4;
  int n    = lane & 15;
  int bh   = blockIdx.y;
  int qt   = blockIdx.x * 8 + wid;
  int q0   = qt * 16;

  const bf16* Qh = Qb + (size_t)bh * S_ * DK_;
  const bf16* Kh = Kb + (size_t)bh * S_ * DK_;
  const bf16* Vh = Vt + (size_t)bh * DK_ * S_;

  // Q A-fragments for dk-steps 0..31 and 32..63 (row m = lane&15)
  const bf16* qrow = Qh + (size_t)(q0 + n) * DK_;
  v16bf aQ[2];
#pragma unroll
  for (int s = 0; s < 2; ++s) {
    v8bf lo = *(const v8bf*)(qrow + s * 32 + half * 8);
    v8bf hi = *(const v8bf*)(qrow + s * 32 + 16 + half * 8);
    aQ[s] = mk16(lo, hi);
  }

  float mrow[8], lrow[8];
  v8f   O[4];
#pragma unroll
  for (int r = 0; r < 8; ++r) { mrow[r] = -1e30f; lrow[r] = 0.0f; }
#pragma unroll
  for (int g = 0; g < 4; ++g) O[g] = zero8();

  const float SC  = 0.125f;                 // 1/sqrt(64)
  const float L2E = 1.4426950408889634f;
  bf16* pw = &pbuf[wid][0];

  for (int kt = 0; kt < S_ / 32; ++kt) {
    // ---- scores for 32 keys: two 16-col C fragments ----
    v8f sc[2];
#pragma unroll
    for (int c = 0; c < 2; ++c) {
      const bf16* krow = Kh + (size_t)(kt * 32 + c * 16 + n) * DK_;
      v8f cacc = zero8();
#pragma unroll
      for (int s = 0; s < 2; ++s) {
        v8bf lo = *(const v8bf*)(krow + s * 32 + half * 8);
        v8bf hi = *(const v8bf*)(krow + s * 32 + 16 + half * 8);
        v16bf bK = mk16(lo, hi);
        cacc = __builtin_amdgcn_wmma_f32_16x16x32_bf16(false, aQ[s], false, bK, (short)0,
                                                       cacc, false, false);
      }
      sc[c] = cacc;
    }

    // ---- online softmax update (row-wise across 16 lanes of each half) ----
    float p0[8], p1[8];
#pragma unroll
    for (int r = 0; r < 8; ++r) {
      float s0 = sc[0][r] * SC;
      float s1 = sc[1][r] * SC;
      float rm = fmaxf(s0, s1);
      rm = fmaxf(rm, __shfl_xor(rm, 1, 32));
      rm = fmaxf(rm, __shfl_xor(rm, 2, 32));
      rm = fmaxf(rm, __shfl_xor(rm, 4, 32));
      rm = fmaxf(rm, __shfl_xor(rm, 8, 32));
      float mn    = fmaxf(mrow[r], rm);
      float alpha = __builtin_amdgcn_exp2f((mrow[r] - mn) * L2E);
      mrow[r] = mn;
      float e0 = __builtin_amdgcn_exp2f((s0 - mn) * L2E);
      float e1 = __builtin_amdgcn_exp2f((s1 - mn) * L2E);
      float rs = e0 + e1;
      rs += __shfl_xor(rs, 1, 32);
      rs += __shfl_xor(rs, 2, 32);
      rs += __shfl_xor(rs, 4, 32);
      rs += __shfl_xor(rs, 8, 32);
      lrow[r] = lrow[r] * alpha + rs;
      p0[r] = e0; p1[r] = e1;
#pragma unroll
      for (int g = 0; g < 4; ++g) O[g][r] *= alpha;
    }

    // ---- C-layout f32 P -> A-layout bf16 via per-wave LDS ----
#pragma unroll
    for (int r = 0; r < 8; ++r) {
      int m = half * 8 + r;
      pw[m * 40 + n]      = (bf16)p0[r];
      pw[m * 40 + 16 + n] = (bf16)p1[r];
    }
    asm volatile("s_wait_dscnt 0" ::: "memory");
    v8bf plo = *(const v8bf*)(pw + n * 40 + half * 8);        // A row m = lane&15
    v8bf phi = *(const v8bf*)(pw + n * 40 + 16 + half * 8);
    v16bf aP = mk16(plo, phi);

    // ---- O += P @ V  (V transposed: [dk][key]) ----
#pragma unroll
    for (int g = 0; g < 4; ++g) {
      const bf16* vrow = Vh + (size_t)(g * 16 + n) * S_ + kt * 32 + half * 8;
      v8bf lo = *(const v8bf*)(vrow);
      v8bf hi = *(const v8bf*)(vrow + 16);
      v16bf bV = mk16(lo, hi);
      O[g] = __builtin_amdgcn_wmma_f32_16x16x32_bf16(false, aP, false, bV, (short)0,
                                                     O[g], false, false);
    }
    asm volatile("" ::: "memory");
  }

  // ---- normalize and write ctx [B,S,E] bf16 ----
  int bb = bh >> 4, h = bh & 15;
#pragma unroll
  for (int r = 0; r < 8; ++r) {
    float inv = 1.0f / lrow[r];
    int m = half * 8 + r;
    size_t row = ((size_t)bb * S_ + (q0 + m)) * E_ + h * DK_;
#pragma unroll
    for (int g = 0; g < 4; ++g)
      ctxb[row + g * 16 + n] = (bf16)(O[g][r] * inv);
  }
}

// ---------------- output projection ----------------
// Each wave computes one 32x32 tile (2x2 WMMA fragments).
// tiles: 128 (M/32) * 32 (E/32) = 4096 ; 8 waves/block -> 512 blocks; fp32 output
__global__ void __launch_bounds__(256)
out_gemm_k(const bf16* __restrict__ ctxb, const bf16* __restrict__ woT,
           const float* __restrict__ bo, float* __restrict__ out) {
  int wid  = threadIdx.x >> 5;
  int lane = threadIdx.x & 31;
  int half = lane >> 4;
  int n    = lane & 15;

  int tile  = blockIdx.x * 8 + wid;
  int mtile = tile & 127;
  int ntile = tile >> 7;

  const bf16* arow0 = ctxb + (size_t)(mtile * 32 + n) * E_;
  const bf16* arow1 = arow0 + (size_t)16 * E_;
  const bf16* brow0 = woT + (size_t)(ntile * 32 + n) * E_;
  const bf16* brow1 = brow0 + (size_t)16 * E_;

  v8f acc[2][2];
#pragma unroll
  for (int mi = 0; mi < 2; ++mi)
#pragma unroll
    for (int ni = 0; ni < 2; ++ni) acc[mi][ni] = zero8();

  for (int k = 0; k < E_; k += 32) {
    v16bf a0 = mk16(*(const v8bf*)(arow0 + k + half * 8),
                    *(const v8bf*)(arow0 + k + 16 + half * 8));
    v16bf a1 = mk16(*(const v8bf*)(arow1 + k + half * 8),
                    *(const v8bf*)(arow1 + k + 16 + half * 8));
    v16bf b0 = mk16(*(const v8bf*)(brow0 + k + half * 8),
                    *(const v8bf*)(brow0 + k + 16 + half * 8));
    v16bf b1 = mk16(*(const v8bf*)(brow1 + k + half * 8),
                    *(const v8bf*)(brow1 + k + 16 + half * 8));
    acc[0][0] = __builtin_amdgcn_wmma_f32_16x16x32_bf16(false, a0, false, b0, (short)0, acc[0][0], false, false);
    acc[0][1] = __builtin_amdgcn_wmma_f32_16x16x32_bf16(false, a0, false, b1, (short)0, acc[0][1], false, false);
    acc[1][0] = __builtin_amdgcn_wmma_f32_16x16x32_bf16(false, a1, false, b0, (short)0, acc[1][0], false, false);
    acc[1][1] = __builtin_amdgcn_wmma_f32_16x16x32_bf16(false, a1, false, b1, (short)0, acc[1][1], false, false);
  }

#pragma unroll
  for (int ni = 0; ni < 2; ++ni) {
    int col = ntile * 32 + ni * 16 + n;
    float bvl = bo[col];
#pragma unroll
    for (int mi = 0; mi < 2; ++mi) {
#pragma unroll
      for (int r = 0; r < 8; ++r) {
        int m = half * 8 + r;
        out[(size_t)(mtile * 32 + mi * 16 + m) * E_ + col] = acc[mi][ni][r] + bvl;
      }
    }
  }
}

// ---------------- launch ----------------
extern "C" void kernel_launch(void* const* d_in, const int* in_sizes, int n_in,
                              void* d_out, int out_size, void* d_ws, size_t ws_size,
                              hipStream_t stream) {
  const float* x  = (const float*)d_in[0];
  const float* Wq = (const float*)d_in[1];
  const float* bq = (const float*)d_in[2];
  const float* Wk = (const float*)d_in[3];
  const float* bk = (const float*)d_in[4];
  const float* Wv = (const float*)d_in[5];
  const float* bv = (const float*)d_in[6];
  const float* Wo = (const float*)d_in[7];
  const float* bo = (const float*)d_in[8];
  float* out = (float*)d_out;

  char* ws = (char*)d_ws;
  bf16* xb  = (bf16*)(ws);                           // 8 MB
  bf16* wqT = (bf16*)(ws + ((size_t)8  << 20));      // 2 MB each
  bf16* wkT = (bf16*)(ws + ((size_t)10 << 20));
  bf16* wvT = (bf16*)(ws + ((size_t)12 << 20));
  bf16* woT = (bf16*)(ws + ((size_t)14 << 20));
  bf16* Qb  = (bf16*)(ws + ((size_t)16 << 20));      // 8 MB
  bf16* Kb  = (bf16*)(ws + ((size_t)24 << 20));      // 8 MB
  bf16* Vt  = (bf16*)(ws + ((size_t)32 << 20));      // 8 MB
  bf16* ctx = (bf16*)(ws + ((size_t)40 << 20));      // 8 MB

  const int NX = B_ * S_ * E_;          // 4194304
  cvt_f32_bf16_k<<<NX / 256, 256, 0, stream>>>(x, xb, NX);
  cvt_wT_k<<<(E_ * E_) / 256, 256, 0, stream>>>(Wq, wqT);
  cvt_wT_k<<<(E_ * E_) / 256, 256, 0, stream>>>(Wk, wkT);
  cvt_wT_k<<<(E_ * E_) / 256, 256, 0, stream>>>(Wv, wvT);
  cvt_wT_k<<<(E_ * E_) / 256, 256, 0, stream>>>(Wo, woT);

  qkv_gemm_k<<<1536, 256, 0, stream>>>(xb, wqT, wkT, wvT, bq, bk, bv, Qb, Kb, Vt);
  attn_k<<<dim3(S_ / 16 / 8, B_ * H_), 256, 0, stream>>>(Qb, Kb, Vt, ctx);
  out_gemm_k<<<512, 256, 0, stream>>>(ctx, woT, bo, out);
}
